// GATModel_71124658421790
// MI455X (gfx1250) — matrix operand order repaired and checked by reading, not compiled, not run on hardware
//
#include <hip/hip_runtime.h>
#include <hip/hip_bf16.h>
#include <math.h>

// Model constants (from reference)
#define IN_DIM   128
#define HID      64
#define HEADS    4
#define FDIM     (HID * HEADS)   // 256
#define NEG_SLOPE 0.2f

typedef __attribute__((ext_vector_type(16))) __bf16 v16bf;
typedef __attribute__((ext_vector_type(8)))  float  v8f;

// ---------------------------------------------------------------------------
// Weight pack: W[k, n] fp32 row-major  ->  Bt[n, k] bf16 (transposed)
// ---------------------------------------------------------------------------
__global__ void pack_wt(const float* __restrict__ W, __bf16* __restrict__ Bt,
                        int K, int Nout) {
  int idx = blockIdx.x * blockDim.x + threadIdx.x;
  if (idx >= K * Nout) return;
  int n = idx / K, k = idx - n * K;
  Bt[(size_t)n * K + k] = (__bf16)W[(size_t)k * Nout + n];
}

// ---------------------------------------------------------------------------
// GEMM: C[M x Nout] = A[M x K] (fp32, cvt->bf16) @ B (packed bf16, transposed)
// One wave -> 16 rows x 64 cols (4 WMMA column tiles).  blockDim = 32*Nout/64.
// M must be a multiple of 16 (N=50000 = 3125*16).
// ---------------------------------------------------------------------------
__global__ void gemm_wmma_bf16(const float* __restrict__ A,
                               const __bf16* __restrict__ Bt,
                               float* __restrict__ C,
                               int K, int Nout) {
  const int wave = threadIdx.x >> 5;
  const int lane = threadIdx.x & 31;
  const int half = lane >> 4;           // 0: lanes 0-15, 1: lanes 16-31
  const int l    = lane & 15;
  const int rowBase = blockIdx.x * 16;
  const int ncol0   = wave * 64;
  if (ncol0 >= Nout) return;

  const float* arow = A + (size_t)(rowBase + l) * K;

  v8f acc[4];
  acc[0] = (v8f){}; acc[1] = (v8f){}; acc[2] = (v8f){}; acc[3] = (v8f){};

  for (int k0 = 0; k0 < K; k0 += 32) {
    // A 16x32 bf16 layout: lanes0-15 hold K = k0+0..7 and k0+16..23,
    //                      lanes16-31 hold K = k0+8..15 and k0+24..31
    v16bf a;
#pragma unroll
    for (int i = 0; i < 8; ++i) a[i]     = (__bf16)arow[k0 + half * 8 + i];
#pragma unroll
    for (int i = 0; i < 8; ++i) a[8 + i] = (__bf16)arow[k0 + 16 + half * 8 + i];

#pragma unroll
    for (int nt = 0; nt < 4; ++nt) {
      // B 32x16 bf16 layout: lane = column; lanes0-15 K=k0+0..7 and k0+16..23,
      //                      lanes16-31 K=k0+8..15 and k0+24..31 (same as A)
      const __bf16* bcol = Bt + (size_t)(ncol0 + nt * 16 + l) * K + k0 + half * 8;
      v16bf b;
#pragma unroll
      for (int i = 0; i < 16; ++i) b[i] = bcol[i < 8 ? i : i + 8];
      acc[nt] = __builtin_amdgcn_wmma_f32_16x16x32_bf16(
          false, a, false, b, (short)0, acc[nt], false, false);
    }
  }

  // C/D layout: VGPR r -> row (r + 8*half), col = lane%16
#pragma unroll
  for (int nt = 0; nt < 4; ++nt) {
    const int col = ncol0 + nt * 16 + l;
    float* cp = C + (size_t)(rowBase + half * 8) * Nout + col;
#pragma unroll
    for (int r = 0; r < 8; ++r)
      cp[(size_t)r * Nout] = acc[nt][r];
  }
}

// ---------------------------------------------------------------------------
// CSR-by-destination build (done once, reused by 3 GAT layers + GCN)
// ---------------------------------------------------------------------------
__global__ void csr_count(const int* __restrict__ ei, int E, int E2,
                          int* __restrict__ cnt) {
  int e = blockIdx.x * blockDim.x + threadIdx.x;
  if (e >= E2) return;
  int d = (e < E) ? ei[E + e] : (e - E);
  atomicAdd(&cnt[d], 1);
}

// single-block scan: rowstart[0..N] from per-row counts
__global__ void csr_scan(const int* __restrict__ cnt, int* __restrict__ rowstart,
                         int N) {
  __shared__ int part[1024];
  const int t = threadIdx.x;
  const int chunk = (N + 1023) / 1024;
  const int beg = t * chunk;
  const int end = min(beg + chunk, N);
  int s = 0;
  for (int i = beg; i < end; ++i) s += cnt[i];
  part[t] = s;
  __syncthreads();
  // Hillis-Steele inclusive scan over 1024 partials
  for (int off = 1; off < 1024; off <<= 1) {
    int v = (t >= off) ? part[t - off] : 0;
    __syncthreads();
    part[t] += v;
    __syncthreads();
  }
  int run = (t == 0) ? 0 : part[t - 1];     // exclusive prefix of this chunk
  for (int i = beg; i < end; ++i) { rowstart[i] = run; run += cnt[i]; }
  if (end == N && beg < N) rowstart[N] = run;   // total = E2
}

__global__ void csr_fill(const int* __restrict__ ei, int E, int E2,
                         const int* __restrict__ rowstart,
                         int* __restrict__ cursor, int* __restrict__ csrc) {
  int e = blockIdx.x * blockDim.x + threadIdx.x;
  if (e >= E2) return;
  int s, d;
  if (e < E) { s = ei[e]; d = ei[E + e]; } else { s = e - E; d = e - E; }
  int pos = atomicAdd(&cursor[d], 1);
  csrc[rowstart[d] + pos] = s;
}

__global__ void deg_inv(const int* __restrict__ rowstart, float* __restrict__ dinv,
                        int N) {
  int n = blockIdx.x * blockDim.x + threadIdx.x;
  if (n >= N) return;
  dinv[n] = rsqrtf((float)max(rowstart[n + 1] - rowstart[n], 1));
}

// ---------------------------------------------------------------------------
// Attention per-node dots: as_n/ad_n [N, HEADS]
// ---------------------------------------------------------------------------
__global__ void att_node(const float* __restrict__ hlin,
                         const float* __restrict__ a_s,
                         const float* __restrict__ a_d,
                         float* __restrict__ as_n, float* __restrict__ ad_n,
                         int N) {
  int idx = blockIdx.x * blockDim.x + threadIdx.x;
  if (idx >= N * HEADS) return;
  int n = idx >> 2, hd = idx & 3;
  const float* hp = hlin + (size_t)n * FDIM + hd * HID;
  float ss = 0.f, sd = 0.f;
#pragma unroll 8
  for (int c = 0; c < HID; ++c) {
    float v = hp[c];
    ss += v * a_s[hd * HID + c];
    sd += v * a_d[hd * HID + c];
  }
  as_n[idx] = ss; ad_n[idx] = sd;
}

// ---------------------------------------------------------------------------
// Fused GAT aggregation (gather form): one wave per (dst, head).
// Online softmax over the dst's in-edges while accumulating the weighted
// feature sum in registers (lane holds 2 features). Fuses logits, softmax,
// message aggregation, bias and ELU -> zero float atomics, one edge pass.
// ---------------------------------------------------------------------------
__global__ void gat_gather(const float* __restrict__ hlin,
                           const float* __restrict__ as_n,
                           const float* __restrict__ ad_n,
                           const int* __restrict__ rowstart,
                           const int* __restrict__ csrc,
                           const float* __restrict__ bias,
                           float* __restrict__ out, int N) {
  const int wid  = (blockIdx.x * blockDim.x + threadIdx.x) >> 5;
  const int lane = threadIdx.x & 31;
  if (wid >= N * HEADS) return;
  const int n = wid >> 2, hd = wid & 3;
  const int r0 = rowstart[n], r1 = rowstart[n + 1];
  const float adv = ad_n[n * HEADS + hd];
  const int f = hd * HID + lane * 2;

  float m = -3.4e38f, l = 0.f;
  float accx = 0.f, accy = 0.f;
  for (int i = r0; i < r1; ++i) {
    const int s = csrc[i];                       // uniform across wave -> broadcast
    float logit = as_n[s * HEADS + hd] + adv;
    logit = logit > 0.f ? logit : NEG_SLOPE * logit;   // leaky relu
    const float nm = fmaxf(m, logit);
    const float sc = __expf(m - nm);
    const float p  = __expf(logit - nm);
    const float2 hv = *(const float2*)(hlin + (size_t)s * FDIM + f);
    l    = l * sc + p;
    accx = accx * sc + p * hv.x;
    accy = accy * sc + p * hv.y;
    m = nm;
  }
  const float inv = 1.f / (l + 1e-16f);
  float vx = accx * inv + bias[f];
  float vy = accy * inv + bias[f + 1];
  vx = vx > 0.f ? vx : expm1f(vx);               // ELU fused
  vy = vy > 0.f ? vy : expm1f(vy);
  float2 o; o.x = vx; o.y = vy;
  *(float2*)(out + (size_t)n * FDIM + f) = o;
}

// ---------------------------------------------------------------------------
// GCN aggregation (gather form): one wave per dst node, lane = 2 features.
// ---------------------------------------------------------------------------
__global__ void gcn_gather(const float* __restrict__ hg,
                           const float* __restrict__ dinv,
                           const int* __restrict__ rowstart,
                           const int* __restrict__ csrc,
                           float* __restrict__ outg, int N) {
  const int wid  = (blockIdx.x * blockDim.x + threadIdx.x) >> 5;
  const int lane = threadIdx.x & 31;
  if (wid >= N) return;
  const int r0 = rowstart[wid], r1 = rowstart[wid + 1];
  const float dd = dinv[wid];
  const int f = lane * 2;
  float accx = 0.f, accy = 0.f;
  for (int i = r0; i < r1; ++i) {
    const int s = csrc[i];
    const float w = dinv[s] * dd;
    const float2 hv = *(const float2*)(hg + (size_t)s * HID + f);
    accx += hv.x * w;
    accy += hv.y * w;
  }
  float2 o; o.x = accx; o.y = accy;
  *(float2*)(outg + (size_t)wid * HID + f) = o;   // bias bg folded into head
}

// ---------------------------------------------------------------------------
// Global mean pool (sum + counts)
// ---------------------------------------------------------------------------
__global__ void pool_sum(const float* __restrict__ hG,
                         const int* __restrict__ batch, int N,
                         float* __restrict__ pool, float* __restrict__ cnt) {
  long long idx = (long long)blockIdx.x * blockDim.x + threadIdx.x;
  if (idx >= (long long)N * HID) return;
  int n = (int)(idx >> 6), f = (int)idx & 63;
  int b = batch[n];
  atomicAdd(&pool[(size_t)b * HID + f], hG[idx]);
  if (f == 0) atomicAdd(&cnt[b], 1.0f);
}

// ---------------------------------------------------------------------------
// Final MLP + log_softmax : one thread per graph (G = 64). bg folded here.
// ---------------------------------------------------------------------------
__global__ void final_mlp(const float* __restrict__ pool,
                          const float* __restrict__ cnt,
                          const float* __restrict__ bg,
                          const float* __restrict__ Wl1, const float* __restrict__ bl1,
                          const float* __restrict__ Wl2, const float* __restrict__ bl2,
                          float* __restrict__ out, int G) {
  int t = blockIdx.x * blockDim.x + threadIdx.x;
  if (t >= G) return;
  float g[HID];
  float c = fmaxf(cnt[t], 1.0f);
  for (int f = 0; f < HID; ++f)
    g[f] = pool[(size_t)t * HID + f] / c + bg[f];
  float lg0 = bl2[0], lg1 = bl2[1];
  for (int j = 0; j < HID; ++j) {
    float s = bl1[j];
    for (int f = 0; f < HID; ++f) s += g[f] * Wl1[f * HID + j];
    s = s > 0.f ? s : expm1f(s);
    lg0 += s * Wl2[j * 2 + 0];
    lg1 += s * Wl2[j * 2 + 1];
  }
  float mx  = fmaxf(lg0, lg1);
  float lse = mx + logf(__expf(lg0 - mx) + __expf(lg1 - mx));
  out[t * 2 + 0] = lg0 - lse;
  out[t * 2 + 1] = lg1 - lse;
}

// ---------------------------------------------------------------------------
// host launcher
// ---------------------------------------------------------------------------
static inline int nblk(long long total, int bs) { return (int)((total + bs - 1) / bs); }

extern "C" void kernel_launch(void* const* d_in, const int* in_sizes, int n_in,
                              void* d_out, int out_size, void* d_ws, size_t ws_size,
                              hipStream_t stream) {
  const float* x      = (const float*)d_in[0];
  const int*   ei     = (const int*)  d_in[1];
  const int*   batch  = (const int*)  d_in[3];
  const float* W1     = (const float*)d_in[4];
  const float* as1    = (const float*)d_in[5];
  const float* ad1    = (const float*)d_in[6];
  const float* b1     = (const float*)d_in[7];
  const float* W2     = (const float*)d_in[8];
  const float* as2    = (const float*)d_in[9];
  const float* ad2    = (const float*)d_in[10];
  const float* b2     = (const float*)d_in[11];
  const float* W3     = (const float*)d_in[12];
  const float* as3    = (const float*)d_in[13];
  const float* ad3    = (const float*)d_in[14];
  const float* b3     = (const float*)d_in[15];
  const float* Wg     = (const float*)d_in[16];
  const float* bg     = (const float*)d_in[17];
  const float* Wl1    = (const float*)d_in[18];
  const float* bl1    = (const float*)d_in[19];
  const float* Wl2    = (const float*)d_in[20];
  const float* bl2    = (const float*)d_in[21];

  const int N  = in_sizes[0] / IN_DIM;
  const int E  = in_sizes[1] / 2;
  const int G  = out_size / 2;
  const int E2 = E + N;

  // bump allocator on workspace (~120 MB total)
  char* wp = (char*)d_ws;
  auto alloc = [&](size_t bytes) -> void* {
    void* p = (void*)wp;
    wp += (bytes + 255) & ~(size_t)255;
    return p;
  };
  float*    P       = (float*)alloc((size_t)N * FDIM * 4);   // layer io (ping)
  float*    Q       = (float*)alloc((size_t)N * FDIM * 4);   // linear out (pong)
  float*    as_n    = (float*)alloc((size_t)N * HEADS * 4);
  float*    ad_n    = (float*)alloc((size_t)N * HEADS * 4);
  int*      rowstart= (int*)  alloc((size_t)(N + 1) * 4);
  int*      cursor  = (int*)  alloc((size_t)N * 4);          // counts, then fill cursor
  int*      csrc    = (int*)  alloc((size_t)E2 * 4);         // src per dst-sorted edge
  float*    dinv    = (float*)alloc((size_t)N * 4);
  float*    hg      = (float*)alloc((size_t)N * HID * 4);
  float*    outg    = (float*)alloc((size_t)N * HID * 4);
  float*    pool    = (float*)alloc((size_t)G * HID * 4);
  float*    cnt     = (float*)alloc((size_t)G * 4);
  __bf16*   W1t     = (__bf16*)alloc((size_t)IN_DIM * FDIM * 2);
  __bf16*   W2t     = (__bf16*)alloc((size_t)FDIM * FDIM * 2);
  __bf16*   W3t     = (__bf16*)alloc((size_t)FDIM * FDIM * 2);
  __bf16*   Wgt     = (__bf16*)alloc((size_t)FDIM * HID * 2);

  // ---- pack weights (bf16, transposed for WMMA B loads) ----
  pack_wt<<<nblk(IN_DIM * FDIM, 256), 256, 0, stream>>>(W1, W1t, IN_DIM, FDIM);
  pack_wt<<<nblk(FDIM * FDIM,   256), 256, 0, stream>>>(W2, W2t, FDIM,   FDIM);
  pack_wt<<<nblk(FDIM * FDIM,   256), 256, 0, stream>>>(W3, W3t, FDIM,   FDIM);
  pack_wt<<<nblk(FDIM * HID,    256), 256, 0, stream>>>(Wg, Wgt, FDIM,   HID);

  // ---- build dst-CSR once (reused by all message-passing layers) ----
  hipMemsetAsync(cursor, 0, (size_t)N * 4, stream);
  csr_count<<<nblk(E2, 256), 256, 0, stream>>>(ei, E, E2, cursor);
  csr_scan<<<1, 1024, 0, stream>>>(cursor, rowstart, N);
  hipMemsetAsync(cursor, 0, (size_t)N * 4, stream);
  csr_fill<<<nblk(E2, 256), 256, 0, stream>>>(ei, E, E2, rowstart, cursor, csrc);
  deg_inv<<<nblk(N, 256), 256, 0, stream>>>(rowstart, dinv, N);

  const int rowTiles = N / 16;                  // 50000 / 16 = 3125, exact

  // ---- 3x GAT layers ----
  const float*  ain[3] = { x,   P,   P   };
  const __bf16* wt [3] = { W1t, W2t, W3t };
  const int     kk [3] = { IN_DIM, FDIM, FDIM };
  const float*  asv[3] = { as1, as2, as3 };
  const float*  adv[3] = { ad1, ad2, ad3 };
  const float*  bv [3] = { b1,  b2,  b3  };

  for (int L = 0; L < 3; ++L) {
    gemm_wmma_bf16<<<rowTiles, 128, 0, stream>>>(ain[L], wt[L], Q, kk[L], FDIM);
    att_node<<<nblk((long long)N * HEADS, 256), 256, 0, stream>>>(Q, asv[L], adv[L],
                                                                  as_n, ad_n, N);
    // 8 waves/block, one wave per (dst, head)
    gat_gather<<<nblk((long long)N * HEADS * 32, 256), 256, 0, stream>>>(
        Q, as_n, ad_n, rowstart, csrc, bv[L], P, N);
  }

  // ---- GCN layer ----
  gemm_wmma_bf16<<<rowTiles, 32, 0, stream>>>(P, Wgt, hg, FDIM, HID);
  gcn_gather<<<nblk((long long)N * 32, 256), 256, 0, stream>>>(
      hg, dinv, rowstart, csrc, outg, N);

  // ---- mean pool + MLP head (bg folded into pooled mean) ----
  hipMemsetAsync(pool, 0, (size_t)G * HID * 4, stream);
  hipMemsetAsync(cnt,  0, (size_t)G * 4, stream);
  pool_sum<<<nblk((long long)N * HID, 256), 256, 0, stream>>>(outg, batch, N, pool, cnt);
  final_mlp<<<1, ((G + 63) / 64) * 64, 0, stream>>>(pool, cnt, bg, Wl1, bl1, Wl2, bl2,
                                                    (float*)d_out, G);
}
